// GPT_20298015441102
// MI455X (gfx1250) — matrix operand order repaired and checked by reading, not compiled, not run on hardware
//
#include <hip/hip_runtime.h>
#include <hip/hip_bf16.h>
#include <math.h>

// ---------------- problem dimensions (fixed by reference) ----------------
#define DIM_N_TOK 4096   // B*T = 2*2048
#define DIM_C     1024
#define DIM_E     64
#define DIM_HE    512
#define DIM_HS    1024
#define TOPK      8
#define CAP       1024   // 2*N*K/E

// ---------------- types ----------------
typedef __bf16 bf16;
typedef __attribute__((ext_vector_type(16))) __bf16 v16bf;
typedef __attribute__((ext_vector_type(8)))  __bf16 bf16x8;
typedef __attribute__((ext_vector_type(4)))  __bf16 bf16x4;
typedef __attribute__((ext_vector_type(8)))  float  v8f;
typedef __attribute__((ext_vector_type(4)))  float  f32x4;
typedef __attribute__((ext_vector_type(4)))  unsigned int u32x4;
typedef __attribute__((ext_vector_type(8)))  int    i32x8;
typedef __attribute__((ext_vector_type(4)))  int    i32x4;

union FragAB { v16bf v; bf16x8 h[2]; };

// ---------------- GEMM tile config ----------------
#define BM 64
#define BN 128
#define BK 64
#define LDK (BK + 8)   // pad: keeps 16B alignment, breaks bank conflicts

// TDM available? (this toolchain: clang-23 / therock-10.0, 6-arg builtin)
#if defined(__HIP_DEVICE_COMPILE__) && __has_builtin(__builtin_amdgcn_tensor_load_to_lds) && __has_builtin(__builtin_amdgcn_s_wait_tensorcnt)
#define USE_TDM 1
#else
#define USE_TDM 0
#endif

// =====================================================================
// Generic bf16 WMMA GEMM:  Out[z] = A[z] (MxK) * B[z] (KxN)
//   A_MODE 0: A fp32 row-major           (converted to bf16 in LDS)
//   A_MODE 1: A fp32 row-gathered via gidx[z*M + row] (token<0 -> zero row)
//   A_MODE 2: A bf16 row-major (direct; staged via TDM tensor_load_to_lds)
//   OUT_BF16: store bf16, else fp32
// Requires M%BM==0, N%BN==0, K%BK==0 (true for all uses here).
// =====================================================================
template <int A_MODE, bool OUT_BF16>
__global__ __launch_bounds__(256) void gemm_bf16_kernel(
    const void* __restrict__ Aptr, const float* __restrict__ Bptr,
    void* __restrict__ Optr, const int* __restrict__ gidx,
    int M, int N, int K, long aBatch, long bBatch, long oBatch)
{
    __shared__ __align__(16) bf16 As[BM][LDK];
    __shared__ __align__(16) bf16 Bs[BN][LDK];   // stored column-major: Bs[n][k]

    const int tid  = threadIdx.x;
    const int z    = blockIdx.z;
    const int row0 = blockIdx.y * BM;
    const int col0 = blockIdx.x * BN;

    const float* Af = (const float*)Aptr + (long)z * aBatch;
    const bf16*  Ab = (const bf16*)Aptr + (long)z * aBatch;
    const float* Bf = Bptr + (long)z * bBatch;
    const int*   gt = (A_MODE == 1) ? (gidx + (long)z * M) : nullptr;

    const int wave = tid >> 5;
    const int lane = tid & 31;
    const int wm   = wave & 1;   // 2 waves along M
    const int wn   = wave >> 1;  // 4 waves along N
    const int l15  = lane & 15;
    const int lh   = lane >> 4;

    v8f acc[2][2] = {};

    for (int k0 = 0; k0 < K; k0 += BK) {
        __syncthreads();   // previous slab's LDS reads complete before refill

        // ---- stage A tile (BM x BK) into LDS as bf16, row-major ----
        if (A_MODE == 2) {
#if USE_TDM
            // Async DMA: 2D bf16 tile, row stride K elems, LDS rows padded to
            // LDK (pad 4 DWORDs every 32 DWORDs). One op per block per slab.
            if (wave == 0) {
                unsigned long long ga =
                    (unsigned long long)(uintptr_t)(Ab + (long)row0 * K + k0);
                unsigned lds = (unsigned)(uintptr_t)&As[0][0]; // addr[31:0] = LDS offset
                u32x4 g0 = {
                    1u,                                        // count=1 (user D#)
                    lds,                                       // lds_addr
                    (unsigned)ga,                              // global_addr[31:0]
                    (unsigned)((ga >> 32) & 0x01FFFFFFu) | (2u << 30) // addr[56:32] | type=2
                };
                i32x8 g1 = {
                    (1 << 16) |                                // data_size = 2 bytes
                    (1 << 20) |                                // pad_enable
                    (4 << 22) |                                // pad_interval: 32 DWORDs
                    (3 << 25),                                 // pad_amount: 4 DWORDs
                    (K & 0xFFFF) << 16,                        // tensor_dim0[15:0]
                    (K >> 16) | ((M & 0xFFFF) << 16),          // tensor_dim0 hi | tensor_dim1 lo
                    (M >> 16) | (BK << 16),                    // tensor_dim1 hi | tile_dim0=64
                    BM,                                        // tile_dim1=64, tile_dim2=0
                    K,                                         // tensor_dim0_stride (elems)
                    0, 0                                       // dim1 stride unused (2D)
                };
                i32x4 gz4 = {0, 0, 0, 0};
                i32x8 gz8 = {0, 0, 0, 0, 0, 0, 0, 0};
                __builtin_amdgcn_tensor_load_to_lds(g0, g1, gz4, gz4, gz8, 0);
            }
#else
            #pragma unroll
            for (int it = 0; it < 2; ++it) {           // 512 chunks of 8 bf16
                int chunk = tid + it * 256;
                int m  = chunk >> 3;
                int kc = (chunk & 7) << 3;
                *(bf16x8*)&As[m][kc] =
                    *(const bf16x8*)(Ab + (long)(row0 + m) * K + k0 + kc);
            }
#endif
        } else {
            #pragma unroll
            for (int it = 0; it < 4; ++it) {           // 1024 chunks of 4 f32
                int chunk = tid + it * 256;
                int m  = chunk >> 4;
                int kc = (chunk & 15) << 2;
                f32x4 v = {0.f, 0.f, 0.f, 0.f};
                if (A_MODE == 1) {
                    int t = gt[row0 + m];
                    if (t >= 0)
                        v = *(const f32x4*)(Af + (long)t * K + k0 + kc);
                } else {
                    v = *(const f32x4*)(Af + (long)(row0 + m) * K + k0 + kc);
                }
                As[m][kc + 0] = (bf16)v[0];
                As[m][kc + 1] = (bf16)v[1];
                As[m][kc + 2] = (bf16)v[2];
                As[m][kc + 3] = (bf16)v[3];
            }
        }

        // ---- stage B tile (BK x BN) into LDS as bf16, transposed ----
        #pragma unroll
        for (int it = 0; it < 8; ++it) {               // 2048 chunks of 4 f32
            int chunk = tid + it * 256;
            int k  = chunk >> 5;
            int nc = (chunk & 31) << 2;
            f32x4 v = *(const f32x4*)(Bf + (long)(k0 + k) * N + col0 + nc);
            Bs[nc + 0][k] = (bf16)v[0];
            Bs[nc + 1][k] = (bf16)v[1];
            Bs[nc + 2][k] = (bf16)v[2];
            Bs[nc + 3][k] = (bf16)v[3];
        }

        // ---- prefetch next B slab: 256 threads x 128B lines = full tile ----
        if (k0 + BK < K) {
            int pr = tid >> 2;            // 64 rows
            int pl = (tid & 3) << 5;      // 4 x 128B lines per row
            __builtin_prefetch(Bf + (long)(k0 + BK + pr) * N + col0 + pl, 0, 3);
        }

#if USE_TDM
        if (A_MODE == 2 && wave == 0)
            __builtin_amdgcn_s_wait_tensorcnt(0);   // A tile DMA complete
#endif
        __syncthreads();

        // ---- 2 WMMA K-steps of 32 over this BK=64 slab ----
        #pragma unroll
        for (int ks = 0; ks < BK; ks += 32) {
            FragAB a[2], b[2];
            #pragma unroll
            for (int i = 0; i < 2; ++i) {
                int r = wm * 32 + i * 16 + l15;
                // 16-bit A 16x32: lanes0-15 K=0..7,16..23 ; lanes16-31 K=8..15,24..31
                a[i].h[0] = *(const bf16x8*)&As[r][ks + lh * 8];
                a[i].h[1] = *(const bf16x8*)&As[r][ks + 16 + lh * 8];
            }
            #pragma unroll
            for (int j = 0; j < 2; ++j) {
                int c = wn * 32 + j * 16 + l15;
                // 16-bit B 32x16: lanes0-15 K=0..7,16..23 ; lanes16-31 K=8..15,24..31
                b[j].h[0] = *(const bf16x8*)&Bs[c][ks + lh * 8];
                b[j].h[1] = *(const bf16x8*)&Bs[c][ks + 16 + lh * 8];
            }
            #pragma unroll
            for (int i = 0; i < 2; ++i)
                #pragma unroll
                for (int j = 0; j < 2; ++j)
                    acc[i][j] = __builtin_amdgcn_wmma_f32_16x16x32_bf16(
                        false, a[i].v, false, b[j].v,
                        (short)0, acc[i][j], false, false);
        }
    }

    // ---- store D: lanes0-15 M=r, lanes16-31 M=8+r ; N = lane&15 ----
    float* Of = (float*)Optr + (long)z * oBatch;
    bf16*  Ob = (bf16*)Optr + (long)z * oBatch;
    #pragma unroll
    for (int i = 0; i < 2; ++i)
        #pragma unroll
        for (int j = 0; j < 2; ++j)
            #pragma unroll
            for (int r = 0; r < 8; ++r) {
                int  row = row0 + wm * 32 + i * 16 + lh * 8 + r;
                int  col = col0 + wn * 32 + j * 16 + l15;
                long off = (long)row * N + col;
                float val = acc[i][j][r];
                if (OUT_BF16) Ob[off] = (bf16)val;
                else          Of[off] = val;
            }
}

// =====================================================================
// Router: scores = sigmoid(x @ W), top-8 (+bias key), normalized weights,
// atomic slot assignment (order-independent for the final sum).
// =====================================================================
__global__ __launch_bounds__(64) void router_topk_kernel(
    const float* __restrict__ x, const float* __restrict__ W,
    const float* __restrict__ bias,
    int* __restrict__ counts, int* __restrict__ slot2tok,
    int* __restrict__ tok2e, int* __restrict__ tok2pos,
    float* __restrict__ tok2w)
{
    __shared__ float xs[DIM_C];
    __shared__ float sc[DIM_E];
    const int t = blockIdx.x;
    const int e = threadIdx.x;
    for (int c = e; c < DIM_C; c += 64) xs[c] = x[(long)t * DIM_C + c];
    __syncthreads();
    float s = 0.f;
    for (int c = 0; c < DIM_C; ++c) s += xs[c] * W[(long)c * DIM_E + e];
    sc[e] = 1.0f / (1.0f + __expf(-s));
    __syncthreads();
    if (e == 0) {
        float keys[DIM_E];
        for (int i = 0; i < DIM_E; ++i) keys[i] = sc[i] + bias[i];
        int sel[TOPK];
        float wsum = 0.f;
        for (int k = 0; k < TOPK; ++k) {
            float best = -1e30f; int bi = 0;
            for (int i = 0; i < DIM_E; ++i)
                if (keys[i] > best) { best = keys[i]; bi = i; }
            sel[k] = bi; keys[bi] = -1e30f; wsum += sc[bi];
        }
        const float inv = 1.0f / wsum;
        for (int k = 0; k < TOPK; ++k) {
            int ex  = sel[k];
            int pos = atomicAdd(&counts[ex], 1);
            tok2e[t * TOPK + k] = ex;
            tok2w[t * TOPK + k] = sc[ex] * inv;
            if (pos < CAP) {
                slot2tok[(long)ex * CAP + pos] = t;
                tok2pos[t * TOPK + k] = pos;
            } else {
                tok2pos[t * TOPK + k] = -1;   // capacity overflow: drop
            }
        }
    }
}

// =====================================================================
// SwiGLU elementwise:  h[r][c] = silu(up[r][half+c]) * up[r][c]
// =====================================================================
__global__ void swiglu_kernel(const bf16* __restrict__ up, bf16* __restrict__ h,
                              long total, int half)
{
    long idx = (long)blockIdx.x * blockDim.x + threadIdx.x;
    if (idx >= total) return;
    long r = idx / half;
    int  c = (int)(idx - r * half);
    float y = (float)up[r * (2 * half) + c];
    float g = (float)up[r * (2 * half) + half + c];
    float sg = g / (1.0f + __expf(-g));
    h[idx] = (bf16)(sg * y);
}

// =====================================================================
// Combine: out[t] += sum_k w_k * oe[e_k, pos_k]   (deterministic gather)
// =====================================================================
__global__ __launch_bounds__(256) void combine_kernel(
    float* __restrict__ out, const bf16* __restrict__ oe,
    const int* __restrict__ tok2e, const int* __restrict__ tok2pos,
    const float* __restrict__ tok2w)
{
    const int t  = blockIdx.x;
    const int c0 = threadIdx.x * 4;           // 256 threads * 4 = C
    f32x4 acc = *(const f32x4*)(out + (long)t * DIM_C + c0);
    #pragma unroll
    for (int k = 0; k < TOPK; ++k) {
        int e   = tok2e[t * TOPK + k];
        int pos = tok2pos[t * TOPK + k];
        if (pos < 0) continue;
        float w = tok2w[t * TOPK + k];
        bf16x4 v = *(const bf16x4*)(oe + ((long)e * CAP + pos) * DIM_C + c0);
        acc[0] += w * (float)v[0];
        acc[1] += w * (float)v[1];
        acc[2] += w * (float)v[2];
        acc[3] += w * (float)v[3];
    }
    *(f32x4*)(out + (long)t * DIM_C + c0) = acc;
}

// =====================================================================
extern "C" void kernel_launch(void* const* d_in, const int* in_sizes, int n_in,
                              void* d_out, int out_size, void* d_ws, size_t ws_size,
                              hipStream_t stream)
{
    const float* x           = (const float*)d_in[0];
    const float* router_w    = (const float*)d_in[1];
    const float* bias        = (const float*)d_in[2];
    const float* shared_up   = (const float*)d_in[3];
    const float* shared_down = (const float*)d_in[4];
    const float* expert_up   = (const float*)d_in[5];
    const float* expert_down = (const float*)d_in[6];
    (void)in_sizes; (void)n_in; (void)out_size; (void)ws_size;

    // ---------------- workspace layout ----------------
    char*  ws  = (char*)d_ws;
    size_t off = 0;
    auto take = [&](size_t bytes) -> char* {
        char* p = ws + off;
        off = (off + bytes + 255) & ~(size_t)255;
        return p;
    };
    int*   counts   = (int*)  take((size_t)DIM_E * sizeof(int));
    int*   slot2tok = (int*)  take((size_t)DIM_E * CAP * sizeof(int));
    int*   tok2e    = (int*)  take((size_t)DIM_N_TOK * TOPK * sizeof(int));
    int*   tok2pos  = (int*)  take((size_t)DIM_N_TOK * TOPK * sizeof(int));
    float* tok2w    = (float*)take((size_t)DIM_N_TOK * TOPK * sizeof(float));
    bf16*  hs       = (bf16*) take((size_t)DIM_N_TOK * DIM_HS * 2);        //  8 MB
    bf16*  up_s     = (bf16*) take((size_t)DIM_N_TOK * 2 * DIM_HS * 2);    // 16 MB
    bf16*  h_e      = (bf16*) take((size_t)DIM_E * CAP * DIM_HE * 2);      // 64 MB
    bf16*  up_e     = (bf16*) take((size_t)DIM_E * CAP * 2 * DIM_HE * 2);  // 128 MB
    bf16*  oe       = up_e;   // reuse after swiglu consumed up_e

    (void)hipMemsetAsync(counts, 0, (size_t)DIM_E * sizeof(int), stream);
    (void)hipMemsetAsync(slot2tok, 0xFF, (size_t)DIM_E * CAP * sizeof(int), stream);

    // 1) routing + top-k + slot assignment
    router_topk_kernel<<<DIM_N_TOK, 64, 0, stream>>>(
        x, router_w, bias, counts, slot2tok, tok2e, tok2pos, tok2w);

    // 2) shared up:  (4096 x 2048) = x (4096x1024) @ shared_up (1024x2048)
    {
        dim3 g((2 * DIM_HS) / BN, DIM_N_TOK / BM, 1);
        gemm_bf16_kernel<0, true><<<g, 256, 0, stream>>>(
            x, shared_up, up_s, nullptr, DIM_N_TOK, 2 * DIM_HS, DIM_C, 0, 0, 0);
    }
    // 3) shared SwiGLU
    {
        long total = (long)DIM_N_TOK * DIM_HS;
        swiglu_kernel<<<(unsigned)((total + 255) / 256), 256, 0, stream>>>(
            up_s, hs, total, DIM_HS);
    }
    // 4) shared down -> d_out (fp32 base; writes every element; A via TDM)
    {
        dim3 g(DIM_C / BN, DIM_N_TOK / BM, 1);
        gemm_bf16_kernel<2, false><<<g, 256, 0, stream>>>(
            hs, shared_down, d_out, nullptr, DIM_N_TOK, DIM_C, DIM_HS, 0, 0, 0);
    }
    // 5) grouped expert up with fused gather: per expert (1024 x 1024 x K=1024)
    {
        dim3 g((2 * DIM_HE) / BN, CAP / BM, DIM_E);
        gemm_bf16_kernel<1, true><<<g, 256, 0, stream>>>(
            x, expert_up, up_e, slot2tok, CAP, 2 * DIM_HE, DIM_C,
            /*aBatch*/0, (long)DIM_C * 2 * DIM_HE, (long)CAP * 2 * DIM_HE);
    }
    // 6) expert SwiGLU
    {
        long total = (long)DIM_E * CAP * DIM_HE;
        swiglu_kernel<<<(unsigned)((total + 255) / 256), 256, 0, stream>>>(
            up_e, h_e, total, DIM_HE);
    }
    // 7) grouped expert down: per expert (1024 x 1024 x K=512) -> oe (bf16, A via TDM)
    {
        dim3 g(DIM_C / BN, CAP / BM, DIM_E);
        gemm_bf16_kernel<2, true><<<g, 256, 0, stream>>>(
            h_e, expert_down, oe, nullptr, CAP, DIM_C, DIM_HE,
            (long)CAP * DIM_HE, (long)DIM_HE * DIM_C, (long)CAP * DIM_C);
    }
    // 8) weighted combine into d_out
    combine_kernel<<<DIM_N_TOK, 256, 0, stream>>>(
        (float*)d_out, oe, tok2e, tok2pos, tok2w);
}
